// DeformableHeadLayer_2327872274643
// MI455X (gfx1250) — matrix operand order repaired and checked by reading, not compiled, not run on hardware
//
#include <hip/hip_runtime.h>
#include <math.h>

// ---------------------------------------------------------------------------
// DeformableHeadLayer on MI455X (gfx1250, wave32)
//  - All six GEMMs on v_wmma_f32_16x16x32_bf16 (bf16 in, f32 accumulate)
//  - GEMM operands pre-packed in bf16 fragment layout: the hot loop is
//    contiguous 32B loads + 4 WMMAs per K-step, software-pipelined so the
//    next K-step's loads overlap the current WMMAs.
//  - LN / sampler / FFN1-GEMM emit packed bf16 directly for their consumers.
//  - Sampling: wave per (b,q,head), lane = channel -> coalesced gathers that
//    hit the 192MB L2 (value tensor is 109MB).
// ---------------------------------------------------------------------------

typedef __attribute__((ext_vector_type(16))) __bf16 v16bf;
typedef __attribute__((ext_vector_type(8)))  float  v8f;

#define B_      8
#define Q_      2048
#define DIM_    256
#define NH_     8
#define HD_     32
#define LEN_IN_ 13294
#define ROWS_   (B_ * Q_)               // 16384
#define MV_     (B_ * LEN_IN_)          // 106352 (rows of value GEMM)
#define XSZ_    ((size_t)ROWS_ * DIM_)  // 4,194,304 floats

// In-block permutation for packed-A layout: swap bits 3 and 4 of k%32.
// (Own inverse.)  A-fragment: lanes 0-15 read K={0..7,16..23} contiguously,
// lanes 16-31 read K={8..15,24..31} contiguously.
__device__ __forceinline__ int packpos(int kk) {
    return (kk & 7) | ((kk & 8) << 1) | ((kk & 16) >> 1);
}

// ---------------------------------------------------------------------------
// Pack A[M,K] f32 -> bf16 fragment layout: out[(m*(K/32)+kb)*32 + packpos(kk)]
// One thread per (m, kb): 128B contiguous read, 64B contiguous write.
// ---------------------------------------------------------------------------
__global__ __launch_bounds__(256) void pack_a_kernel(
    const float* __restrict__ A, __bf16* __restrict__ out, int M, int K)
{
    int idx = blockIdx.x * blockDim.x + threadIdx.x;      // m*(K/32)+kb
    const int nkb = K >> 5;
    if (idx >= M * nkb) return;
    const float* src = A + (size_t)idx * 32;              // m*K + kb*32
    __bf16 buf[32];
#pragma unroll
    for (int kk = 0; kk < 32; ++kk) buf[packpos(kk)] = (__bf16)src[kk];
    __bf16* dst = out + (size_t)idx * 32;
#pragma unroll
    for (int p = 0; p < 32; ++p) dst[p] = buf[p];
}

// ---------------------------------------------------------------------------
// Pack W[K,N] f32 -> bf16 layout: out[(kb*N+n)*32 + packpos(kk)].
// One thread per (kb, n): strided reads coalesce across n, 64B write.
// ---------------------------------------------------------------------------
__global__ __launch_bounds__(256) void pack_w_kernel(
    const float* __restrict__ W, __bf16* __restrict__ out, int K, int N)
{
    int idx = blockIdx.x * blockDim.x + threadIdx.x;      // kb*N + n
    if (idx >= (K >> 5) * N) return;
    const int kb = idx / N, n = idx - kb * N;
    const float* src = W + (size_t)kb * 32 * N + n;
    __bf16* dst = out + (size_t)idx * 32;
#pragma unroll
    for (int kk = 0; kk < 32; ++kk) dst[packpos(kk)] = (__bf16)src[(size_t)kk * N];
}

// ---------------------------------------------------------------------------
// LayerNorm over last dim (256), emitting PACKED bf16 (GEMM-A layout).
// One wave per row; lane holds 8 consecutive channels; the packed positions
// of those 8 channels are contiguous -> one 16B bf16 store per lane.
// ---------------------------------------------------------------------------
template <bool ADD_POS>
__global__ __launch_bounds__(256) void ln_pack_kernel(
    const float* __restrict__ x, const float* __restrict__ g,
    const float* __restrict__ bb, const float* __restrict__ addend,
    __bf16* __restrict__ out, int rows)
{
    int wave = (int)((blockIdx.x * blockDim.x + threadIdx.x) >> 5);
    int lane = threadIdx.x & 31;
    if (wave >= rows) return;

    const float* xr = x + (size_t)wave * DIM_;
    float v[8];
    float s = 0.f;
#pragma unroll
    for (int e = 0; e < 8; ++e) { v[e] = xr[lane * 8 + e]; s += v[e]; }
#pragma unroll
    for (int o = 16; o > 0; o >>= 1) s += __shfl_xor(s, o, 32);
    float mean = s * (1.f / DIM_);

    float var = 0.f;
#pragma unroll
    for (int e = 0; e < 8; ++e) { float d = v[e] - mean; var += d * d; }
#pragma unroll
    for (int o = 16; o > 0; o >>= 1) var += __shfl_xor(var, o, 32);
    float inv = rsqrtf(var * (1.f / DIM_) + 1e-5f);

    // c = lane*8+e ; block = lane>>2 ; packed base within block:
    static const int pbase[4] = {0, 16, 8, 24};           // packpos((lane&3)*8)
    __bf16* dst = out + (size_t)wave * DIM_ + (lane >> 2) * 32 + pbase[lane & 3];
#pragma unroll
    for (int e = 0; e < 8; ++e) {
        int c = lane * 8 + e;
        float r = (v[e] - mean) * inv * g[c] + bb[c];
        if (ADD_POS) r += addend[(size_t)wave * DIM_ + c];
        dst[e] = (__bf16)r;
    }
}

// ---------------------------------------------------------------------------
// GEMM  C[M,N] = epi(Ap[M,K] @ Wp[K,N] + bias), operands pre-packed bf16.
// Wave tile 16x64: one A fragment drives 4 WMMAs per K-step of 32.
// Software-pipelined: K-step k+1's loads are issued before the WMMAs that
// consume K-step k, so VMEM latency overlaps matrix-pipe latency.
// M%16==0, N%64==0, K%32==0 at every call site; branches wave-uniform so
// EXEC is all-ones at each WMMA.
// PACK_OUT: write result as packed-A bf16 for the next GEMM (FFN1 -> FFN2).
// ---------------------------------------------------------------------------
template <bool HAS_RES, bool HAS_MASK, bool RELU, bool PACK_OUT>
__global__ __launch_bounds__(128) void gemm_bf16_wmma(
    const __bf16* __restrict__ Ap, const __bf16* __restrict__ Wp,
    const float* __restrict__ bias, const float* __restrict__ residual,
    const unsigned char* __restrict__ mask, float* __restrict__ C,
    __bf16* __restrict__ Cp, int M, int N, int K)
{
    const int lane = threadIdx.x & 31;
    const int ntn  = N >> 6;                    // tiles of 64 columns
    const int numTiles = (M >> 4) * ntn;
    int tile = blockIdx.x * 4 + (threadIdx.x >> 5);
    if (tile >= numTiles) return;               // wave-uniform

    const int tm   = tile / ntn;
    const int tn   = tile % ntn;
    const int row  = tm * 16 + (lane & 15);     // A row owned by this lane
    const int col0 = tn * 64 + (lane & 15);     // first of 4 fragment columns
    const int hi   = lane >> 4;

    const __bf16* aP = Ap + (size_t)row * K + hi * 16;
    const __bf16* wP = Wp + (size_t)col0 * 32 + hi * 16;
    const size_t wStep = (size_t)N * 32;        // per K-block advance

    v8f acc0 = {}, acc1 = {}, acc2 = {}, acc3 = {};

    // ---- prologue: load K-step 0 fragments --------------------------------
    v16bf a  = *(const v16bf*)aP;
    v16bf b0 = *(const v16bf*)(wP);
    v16bf b1 = *(const v16bf*)(wP + 16 * 32);
    v16bf b2 = *(const v16bf*)(wP + 32 * 32);
    v16bf b3 = *(const v16bf*)(wP + 48 * 32);

    const int nkb = K >> 5;                     // >= 8 at every call site
    for (int kb = 0; kb < nkb - 1; ++kb) {
        aP += 32;
        wP += wStep;
        // issue next K-step's loads first...
        v16bf an = *(const v16bf*)aP;
        v16bf c0 = *(const v16bf*)(wP);
        v16bf c1 = *(const v16bf*)(wP + 16 * 32);
        v16bf c2 = *(const v16bf*)(wP + 32 * 32);
        v16bf c3 = *(const v16bf*)(wP + 48 * 32);
        __builtin_prefetch(aP + 32, 0, 1);      // global_prefetch_b8

        // ...then consume current fragments (loads overlap WMMA latency)
        acc0 = __builtin_amdgcn_wmma_f32_16x16x32_bf16(false, a, false, b0,
                                                       (short)0, acc0, false, false);
        acc1 = __builtin_amdgcn_wmma_f32_16x16x32_bf16(false, a, false, b1,
                                                       (short)0, acc1, false, false);
        acc2 = __builtin_amdgcn_wmma_f32_16x16x32_bf16(false, a, false, b2,
                                                       (short)0, acc2, false, false);
        acc3 = __builtin_amdgcn_wmma_f32_16x16x32_bf16(false, a, false, b3,
                                                       (short)0, acc3, false, false);
        a = an; b0 = c0; b1 = c1; b2 = c2; b3 = c3;
    }
    // ---- drain: last K-step ----------------------------------------------
    acc0 = __builtin_amdgcn_wmma_f32_16x16x32_bf16(false, a, false, b0,
                                                   (short)0, acc0, false, false);
    acc1 = __builtin_amdgcn_wmma_f32_16x16x32_bf16(false, a, false, b1,
                                                   (short)0, acc1, false, false);
    acc2 = __builtin_amdgcn_wmma_f32_16x16x32_bf16(false, a, false, b2,
                                                   (short)0, acc2, false, false);
    acc3 = __builtin_amdgcn_wmma_f32_16x16x32_bf16(false, a, false, b3,
                                                   (short)0, acc3, false, false);

    // ----- epilogue (straight-line; variant chosen at compile time) --------
    const int mBase = tm * 16 + (hi ? 8 : 0);
    const float bc0 = bias[col0];
    const float bc1 = bias[col0 + 16];
    const float bc2 = bias[col0 + 32];
    const float bc3 = bias[col0 + 48];

    // packed-output positions (PACK_OUT): columns col0+{0,16,32,48}
    const int c    = lane & 15;
    const int pA   = (c < 8) ? c : c + 8;        // packpos(c)
    const int pB   = (c < 8) ? c + 8 : c + 16;   // packpos(16+c)
    const int nbl  = N >> 5;                     // K-blocks of the NEXT gemm

#pragma unroll
    for (int v = 0; v < 8; ++v) {
        const int m = mBase + v;
        float r0 = acc0[v] + bc0;
        float r1 = acc1[v] + bc1;
        float r2 = acc2[v] + bc2;
        float r3 = acc3[v] + bc3;
        if (RELU) {
            r0 = fmaxf(r0, 0.f); r1 = fmaxf(r1, 0.f);
            r2 = fmaxf(r2, 0.f); r3 = fmaxf(r3, 0.f);
        }
        if (PACK_OUT) {
            __bf16* dp = Cp + ((size_t)m * nbl + tn * 2) * 32;
            dp[pA]      = (__bf16)r0;            // col0    -> block tn*2
            dp[pB]      = (__bf16)r1;            // col0+16 -> block tn*2
            dp[32 + pA] = (__bf16)r2;            // col0+32 -> block tn*2+1
            dp[32 + pB] = (__bf16)r3;            // col0+48 -> block tn*2+1
        } else {
            const size_t base = (size_t)m * N + col0;
            if (HAS_RES) {
                r0 += residual[base];
                r1 += residual[base + 16];
                r2 += residual[base + 32];
                r3 += residual[base + 48];
            }
            if (HAS_MASK) {
                if (mask[m]) { r0 = 0.f; r1 = 0.f; r2 = 0.f; r3 = 0.f; }
            }
            C[base]      = r0;
            C[base + 16] = r1;
            C[base + 32] = r2;
            C[base + 48] = r3;
        }
    }
}

// ---------------------------------------------------------------------------
// Per-(b,q,head): softmax over 16 attn logits (in place) and sampling
// locations loc = ref + off / normalizer, written straight into d_out.
// ---------------------------------------------------------------------------
__global__ __launch_bounds__(256) void softmax_loc_kernel(
    const float* __restrict__ offs, float* __restrict__ attn,
    const float* __restrict__ refpt, float* __restrict__ loc)
{
    int idx = blockIdx.x * blockDim.x + threadIdx.x;
    if (idx >= ROWS_ * NH_) return;
    const int h  = idx & 7;
    const int bq = idx >> 3;

    float* ar = attn + (size_t)bq * (NH_ * 16) + h * 16;
    float a[16];
    float mx = -1e30f;
#pragma unroll
    for (int i = 0; i < 16; ++i) { a[i] = ar[i]; mx = fmaxf(mx, a[i]); }
    float sum = 0.f;
#pragma unroll
    for (int i = 0; i < 16; ++i) { a[i] = __expf(a[i] - mx); sum += a[i]; }
    const float inv = 1.f / sum;
#pragma unroll
    for (int i = 0; i < 16; ++i) ar[i] = a[i] * inv;

    const float nrm[4] = {100.f, 50.f, 25.f, 13.f};   // W==H per level
    const float* off = offs + (size_t)bq * DIM_ + h * 32;
    const float* ref = refpt + (size_t)bq * 8;        // (L,2) per (b,q)
    float* lo = loc + ((size_t)bq * NH_ + h) * 32;
#pragma unroll
    for (int l = 0; l < 4; ++l) {
        const float inw = 1.f / nrm[l];
#pragma unroll
        for (int p = 0; p < 4; ++p) {
            const int o = (l * 4 + p) * 2;
            lo[o]     = ref[l * 2]     + off[o]     * inw;
            lo[o + 1] = ref[l * 2 + 1] + off[o + 1] * inw;
        }
    }
}

// ---------------------------------------------------------------------------
// MS-deform-attn core: wave per (b,q,head); lane = channel (HEAD_DIM == 32).
// Coalesced corner gathers (value is L2-resident). Result emitted directly
// in packed-A bf16 layout for the Wout GEMM (one b16 store per lane).
// ---------------------------------------------------------------------------
__global__ __launch_bounds__(256) void sample_kernel(
    const float* __restrict__ value, const float* __restrict__ loc,
    const float* __restrict__ attn, __bf16* __restrict__ outP)
{
    int wave = (int)((blockIdx.x * blockDim.x + threadIdx.x) >> 5);
    int lane = threadIdx.x & 31;
    if (wave >= ROWS_ * NH_) return;
    const int h  = wave & 7;
    const int bq = wave >> 3;
    const int b  = bq >> 11;                    // Q = 2048

    const int HH[4] = {100, 50, 25, 13};
    const int ST[4] = {0, 10000, 12500, 13125};

    const float* lc = loc + (size_t)wave * 32;
    const float* at = attn + (size_t)bq * (NH_ * 16) + h * 16;
    float acc = 0.f;

#pragma unroll
    for (int l = 0; l < 4; ++l) {
        const int Hh = HH[l], Wl = HH[l];
        const float* vb =
            value + ((size_t)b * LEN_IN_ + ST[l]) * DIM_ + h * HD_ + lane;
#pragma unroll
        for (int p = 0; p < 4; ++p) {
            const float xx = lc[(l * 4 + p) * 2]     * (float)Wl - 0.5f;
            const float yy = lc[(l * 4 + p) * 2 + 1] * (float)Hh - 0.5f;
            const float x0f = floorf(xx), y0f = floorf(yy);
            const float wx = xx - x0f, wy = yy - y0f;
            const int x0 = (int)x0f, y0 = (int)y0f;
            const bool xv0 = (x0 >= 0) && (x0 < Wl);
            const bool xv1 = (x0 + 1 >= 0) && (x0 + 1 < Wl);
            const bool yv0 = (y0 >= 0) && (y0 < Hh);
            const bool yv1 = (y0 + 1 >= 0) && (y0 + 1 < Hh);
            float g00 = 0.f, g01 = 0.f, g10 = 0.f, g11 = 0.f;
            if (xv0 && yv0) g00 = vb[(size_t)(y0 * Wl + x0) * DIM_];
            if (xv1 && yv0) g01 = vb[(size_t)(y0 * Wl + x0 + 1) * DIM_];
            if (xv0 && yv1) g10 = vb[(size_t)((y0 + 1) * Wl + x0) * DIM_];
            if (xv1 && yv1) g11 = vb[(size_t)((y0 + 1) * Wl + x0 + 1) * DIM_];
            const float top = g00 * (1.f - wx) + g01 * wx;
            const float bot = g10 * (1.f - wx) + g11 * wx;
            acc += at[l * 4 + p] * (top * (1.f - wy) + bot * wy);
        }
    }
    // packed-A layout: row bq, channel block h, in-block pos = packpos(lane)
    outP[(size_t)bq * DIM_ + h * 32 + packpos(lane)] = (__bf16)acc;
}

// ---------------------------------------------------------------------------
extern "C" void kernel_launch(void* const* d_in, const int* in_sizes, int n_in,
                              void* d_out, int out_size, void* d_ws,
                              size_t ws_size, hipStream_t stream)
{
    (void)in_sizes; (void)n_in; (void)out_size; (void)ws_size;

    const float* input  = (const float*)d_in[0];
    const float* pos    = (const float*)d_in[1];
    const float* refpt  = (const float*)d_in[2];
    const float* source = (const float*)d_in[3];
    // d_in[4] source_shape / d_in[5] level_start: compile-time constants here
    const unsigned char* mask = (const unsigned char*)d_in[6];
    const float* ln1g = (const float*)d_in[7];
    const float* ln1b = (const float*)d_in[8];
    const float* ln2g = (const float*)d_in[9];
    const float* ln2b = (const float*)d_in[10];
    const float* Wv   = (const float*)d_in[11];
    const float* bv   = (const float*)d_in[12];
    const float* Woff = (const float*)d_in[13];
    const float* boff = (const float*)d_in[14];
    const float* Watt = (const float*)d_in[15];
    const float* batt = (const float*)d_in[16];
    const float* Wout = (const float*)d_in[17];
    const float* bout = (const float*)d_in[18];
    const float* W1   = (const float*)d_in[19];
    const float* b1   = (const float*)d_in[20];
    const float* W2   = (const float*)d_in[21];
    const float* b2   = (const float*)d_in[22];

    float* xout = (float*)d_out;        // output x: (B,Q,DIM)
    float* loc  = xout + XSZ_;          // output loc: (B,Q,H,L,P,2)

    // Workspace carve-up (~250 MB). All sections 256B-aligned by construction.
    char* w = (char*)d_ws;
    float*  valb  = (float*)w;  w += (size_t)MV_ * DIM_ * 4;       // 108.9 MB
    float*  offs  = (float*)w;  w += XSZ_ * 4;                     //  16.8 MB
    float*  attn  = (float*)w;  w += (size_t)ROWS_ * NH_ * 16 * 4; //   8.4 MB
    __bf16* srcP  = (__bf16*)w; w += (size_t)MV_ * DIM_ * 2;       //  54.5 MB
    __bf16* qpP   = (__bf16*)w; w += XSZ_ * 2;                     //   8.4 MB
    __bf16* sampP = (__bf16*)w; w += XSZ_ * 2;                     //   8.4 MB
    __bf16* hln2P = (__bf16*)w; w += XSZ_ * 2;                     //   8.4 MB
    __bf16* ffnP  = (__bf16*)w; w += (size_t)ROWS_ * 1024 * 2;     //  33.6 MB
    __bf16* WvP   = (__bf16*)w; w += (size_t)256 * 256 * 2;
    __bf16* WoffP = (__bf16*)w; w += (size_t)256 * 256 * 2;
    __bf16* WattP = (__bf16*)w; w += (size_t)256 * 128 * 2;
    __bf16* WoutP = (__bf16*)w; w += (size_t)256 * 256 * 2;
    __bf16* W1P   = (__bf16*)w; w += (size_t)256 * 1024 * 2;
    __bf16* W2P   = (__bf16*)w; w += (size_t)1024 * 256 * 2;

    // 0) pack weights (tiny) and source (one pass)
    pack_w_kernel<<<(8  * 256 + 255) / 256, 256, 0, stream>>>(Wv,   WvP,   256, 256);
    pack_w_kernel<<<(8  * 256 + 255) / 256, 256, 0, stream>>>(Woff, WoffP, 256, 256);
    pack_w_kernel<<<(8  * 128 + 255) / 256, 256, 0, stream>>>(Watt, WattP, 256, 128);
    pack_w_kernel<<<(8  * 256 + 255) / 256, 256, 0, stream>>>(Wout, WoutP, 256, 256);
    pack_w_kernel<<<(8  * 1024+ 255) / 256, 256, 0, stream>>>(W1,   W1P,   256, 1024);
    pack_w_kernel<<<(32 * 256 + 255) / 256, 256, 0, stream>>>(W2,   W2P,   1024, 256);
    pack_a_kernel<<<(MV_ * 8 + 255) / 256, 256, 0, stream>>>(source, srcP, MV_, 256);

    // 1) qp = LN1(input) + pos  (packed bf16)
    ln_pack_kernel<true><<<ROWS_ / 8, 256, 0, stream>>>(
        input, ln1g, ln1b, pos, qpP, ROWS_);

    // 2) value = mask0(source @ Wv + bv)          M=106352 N=256 K=256
    gemm_bf16_wmma<false, true, false, false>
        <<<((MV_ / 16) * 4 + 3) / 4, 128, 0, stream>>>(
            srcP, WvP, bv, nullptr, mask, valb, nullptr, MV_, 256, 256);
    // 3) offsets = qp @ Woff + boff               M=16384 N=256 K=256
    gemm_bf16_wmma<false, false, false, false>
        <<<(ROWS_ / 16) * 4 / 4, 128, 0, stream>>>(
            qpP, WoffP, boff, nullptr, nullptr, offs, nullptr, ROWS_, 256, 256);
    // 4) attn logits = qp @ Wattn + battn         M=16384 N=128 K=256
    gemm_bf16_wmma<false, false, false, false>
        <<<(ROWS_ / 16) * 2 / 4, 128, 0, stream>>>(
            qpP, WattP, batt, nullptr, nullptr, attn, nullptr, ROWS_, 128, 256);
    // 5) softmax(attn) in place; loc -> d_out second half
    softmax_loc_kernel<<<(ROWS_ * NH_) / 256, 256, 0, stream>>>(
        offs, attn, refpt, loc);
    // 6) bilinear sampling -> packed bf16
    sample_kernel<<<(ROWS_ * NH_ * 32) / 256, 256, 0, stream>>>(
        valb, loc, attn, sampP);
    // 7) x = input + samp @ Wout + bout -> d_out
    gemm_bf16_wmma<true, false, false, false>
        <<<(ROWS_ / 16) * 4 / 4, 128, 0, stream>>>(
            sampP, WoutP, bout, input, nullptr, xout, nullptr, ROWS_, 256, 256);
    // 8) hln2 = LN2(x)  (packed bf16)
    ln_pack_kernel<false><<<ROWS_ / 8, 256, 0, stream>>>(
        xout, ln2g, ln2b, nullptr, hln2P, ROWS_);
    // 9) ffn = relu(hln2 @ W1 + b1) -> packed bf16  M=16384 N=1024 K=256
    gemm_bf16_wmma<false, false, true, true>
        <<<(ROWS_ / 16) * 16 / 4, 128, 0, stream>>>(
            hln2P, W1P, b1, nullptr, nullptr, nullptr, ffnP, ROWS_, 1024, 256);
    // 10) x += ffn @ W2 + b2 -> d_out             M=16384 N=256 K=1024
    gemm_bf16_wmma<true, false, false, false>
        <<<(ROWS_ / 16) * 4 / 4, 128, 0, stream>>>(
            ffnP, W2P, b2, xout, nullptr, xout, nullptr, ROWS_, 256, 1024);
}
